// GCN_16638703305286
// MI455X (gfx1250) — compile-verified, matrix operand order checked by reading, not run on hardware
//
#include <hip/hip_runtime.h>

#define NODES 50000
#define DIM   256
#define HALF  128
#define NEDGE 800000
#define NEE   100000

typedef __attribute__((ext_vector_type(16))) __bf16 v16bf;
typedef __attribute__((ext_vector_type(8)))  float  v8f;

// ---------------- utility kernels ----------------

__global__ void zero_f32(float* __restrict__ p, size_t n) {
    for (size_t i = (size_t)blockIdx.x * blockDim.x + threadIdx.x; i < n;
         i += (size_t)gridDim.x * blockDim.x)
        p[i] = 0.0f;
}

__global__ void deg_kernel(const int* __restrict__ dst, float* __restrict__ deg, int ne) {
    int i = blockIdx.x * blockDim.x + threadIdx.x;
    if (i < ne) atomicAdd(&deg[dst[i]], 1.0f);
}

__global__ void rsqrt_kernel(float* __restrict__ deg, int n) {
    int i = blockIdx.x * blockDim.x + threadIdx.x;
    if (i < n) {
        float d = deg[i];
        deg[i] = d > 0.0f ? rsqrtf(d) : 0.0f;
    }
}

// convert conv_W (L,D,D) -> Wt[l][n][k] bf16, and mlp_W1 (D,HALF) -> W1t[n][k] bf16
__global__ void convert_weights(const float* __restrict__ convW, const float* __restrict__ W1,
                                __bf16* __restrict__ Wt, __bf16* __restrict__ W1t) {
    int i = blockIdx.x * blockDim.x + threadIdx.x;
    if (i < 2 * DIM * DIM) {
        int l = i >> 16;
        int n = (i >> 8) & (DIM - 1);
        int k = i & (DIM - 1);
        Wt[i] = (__bf16)convW[(l << 16) + (k << 8) + n];
    }
    if (i < HALF * DIM) {
        int n = i >> 8;          // 0..127
        int k = i & (DIM - 1);   // 0..255
        W1t[i] = (__bf16)W1[k * HALF + n];
    }
}

// x[i] = relu(x[i] + b[i % 256])   (in place on agg buffer)
__global__ void bias_relu(float* __restrict__ x, const float* __restrict__ b, int total) {
    int i = blockIdx.x * blockDim.x + threadIdx.x;
    if (i < total) {
        float v = x[i] + b[i & (DIM - 1)];
        x[i] = v > 0.0f ? v : 0.0f;
    }
}

// ---------------- WMMA GEMM: H[rows x 256] = X[rows x 256] @ W (Wt is [n][k] bf16) ----------------

__global__ __launch_bounds__(256) void gemm16_bf16(const float* __restrict__ X,
                                                   const __bf16* __restrict__ Wt,
                                                   float* __restrict__ H) {
    __shared__ __bf16 As[16][DIM + 8];   // +8 bf16 pad -> row stride 528B, kills bank conflicts
    const int row0 = blockIdx.x * 16;
    const int tid  = threadIdx.x;

    // stage 16x256 f32 strip of X into LDS as bf16 (16 elems/thread)
    for (int i = tid; i < 16 * DIM; i += 256) {
        int r = i >> 8, k = i & (DIM - 1);
        As[r][k] = (__bf16)X[(size_t)(row0 + r) * DIM + k];
    }
    __syncthreads();

    const int wave = tid >> 5, lane = tid & 31;
    const int m = lane & 15, hi = lane >> 4;

#pragma unroll
    for (int t = 0; t < 2; ++t) {               // 8 waves x 2 tiles = 16 column tiles
        const int ntile = wave * 2 + t;
        const int ncol  = ntile * 16 + m;
        const __bf16* wp = Wt + (size_t)ncol * DIM;
        v8f c = {};
#pragma unroll
        for (int kk = 0; kk < DIM; kk += 32) {
            // 16-bit A 16x32 layout: lanes 0-15 hold K {kk..kk+7, kk+16..kk+23},
            // lanes 16-31 hold K {kk+8..kk+15, kk+24..kk+31}; B mirrors with N<->M.
            const int k0 = kk + hi * 8;
            const int k1 = k0 + 16;
            v16bf a, b;
#pragma unroll
            for (int e2 = 0; e2 < 8; ++e2) {
                a[e2]     = As[m][k0 + e2];
                a[8 + e2] = As[m][k1 + e2];
                b[e2]     = wp[k0 + e2];
                b[8 + e2] = wp[k1 + e2];
            }
            c = __builtin_amdgcn_wmma_f32_16x16x32_bf16(false, a, false, b,
                                                        (short)0, c, false, false);
        }
        // C/D: VGPR r -> row (r + hi*8), col = lane&15
#pragma unroll
        for (int r = 0; r < 8; ++r)
            H[(size_t)(row0 + r + hi * 8) * DIM + ntile * 16 + m] = c[r];
    }
}

// ---------------- edge scatter: agg[dst] += h[src] * dis[src]*dis[dst] ----------------

__global__ __launch_bounds__(256) void scatter_msg(const float* __restrict__ H,
                                                   const int* __restrict__ src,
                                                   const int* __restrict__ dst,
                                                   const float* __restrict__ dis,
                                                   float* __restrict__ agg, int ne) {
    const int wid = blockIdx.x * 8 + (threadIdx.x >> 5);   // one wave per edge
    if (wid >= ne) return;
    const int lane = threadIdx.x & 31;
    const int s = src[wid];
    const int d = dst[wid];
    const float nrm = dis[s] * dis[d];
    const float4* hp = (const float4*)(H + (size_t)s * DIM);
    float* ap = agg + (size_t)d * DIM;
#pragma unroll
    for (int j = 0; j < 2; ++j) {
        const int q = lane + j * 32;       // float4 index 0..63
        float4 v = hp[q];
        const int base = q * 4;
        atomicAdd(ap + base + 0, v.x * nrm);
        atomicAdd(ap + base + 1, v.y * nrm);
        atomicAdd(ap + base + 2, v.z * nrm);
        atomicAdd(ap + base + 3, v.w * nrm);
    }
}

// ---------------- fused link-prediction MLP ----------------
// per block: 16 edges.  had = x[e0]*x[e1]  (16x256, bf16 in LDS)
// hid = relu(had @ W1 + b1) via WMMA (8 waves x one 16x16 tile over N=128)
// score = hid @ W2 + b2 via shfl reduction + LDS atomics.

__global__ __launch_bounds__(256) void mlp_kernel(const float* __restrict__ X,
                                                  const int* __restrict__ edge,
                                                  const __bf16* __restrict__ W1t,
                                                  const float* __restrict__ b1,
                                                  const float* __restrict__ W2,
                                                  const float* __restrict__ b2,
                                                  float* __restrict__ out) {
    __shared__ __bf16 Hs[16][DIM + 8];
    __shared__ float scoreAcc[16];
    const int e0  = blockIdx.x * 16;
    const int tid = threadIdx.x;
    if (tid < 16) scoreAcc[tid] = 0.0f;

    for (int i = tid; i < 16 * DIM; i += 256) {
        int r = i >> 8, k = i & (DIM - 1);
        int e = e0 + r;
        int na = edge[e * 2 + 0];
        int nb = edge[e * 2 + 1];
        float v = X[(size_t)na * DIM + k] * X[(size_t)nb * DIM + k];
        Hs[r][k] = (__bf16)v;
    }
    __syncthreads();

    const int wave = tid >> 5, lane = tid & 31;
    const int m = lane & 15, hi = lane >> 4;
    const int ncol = wave * 16 + m;          // 8 waves cover N = 128
    const __bf16* wp = W1t + (size_t)ncol * DIM;

    v8f c = {};
#pragma unroll
    for (int kk = 0; kk < DIM; kk += 32) {
        const int k0 = kk + hi * 8;
        const int k1 = k0 + 16;
        v16bf a, b;
#pragma unroll
        for (int e2 = 0; e2 < 8; ++e2) {
            a[e2]     = Hs[m][k0 + e2];
            a[8 + e2] = Hs[m][k1 + e2];
            b[e2]     = wp[k0 + e2];
            b[8 + e2] = wp[k1 + e2];
        }
        c = __builtin_amdgcn_wmma_f32_16x16x32_bf16(false, a, false, b,
                                                    (short)0, c, false, false);
    }

    const float bias = b1[ncol];
    const float w2v  = W2[ncol];
#pragma unroll
    for (int r = 0; r < 8; ++r) {
        float v = fmaxf(c[r] + bias, 0.0f) * w2v;   // relu(hid)*W2[n]
#pragma unroll
        for (int off = 1; off < 16; off <<= 1)      // reduce over the 16 columns in tile
            v += __shfl_xor(v, off, 32);
        if (m == 0) atomicAdd(&scoreAcc[r + hi * 8], v);  // ds_add_f32
    }
    __syncthreads();
    if (tid < 16) out[e0 + tid] = scoreAcc[tid] + b2[0];
}

// ---------------- launcher ----------------

extern "C" void kernel_launch(void* const* d_in, const int* in_sizes, int n_in,
                              void* d_out, int out_size, void* d_ws, size_t ws_size,
                              hipStream_t stream) {
    const float* emb   = (const float*)d_in[0];
    const float* convW = (const float*)d_in[1];
    const float* convB = (const float*)d_in[2];
    const float* W1    = (const float*)d_in[3];
    const float* b1    = (const float*)d_in[4];
    const float* W2    = (const float*)d_in[5];
    const float* b2    = (const float*)d_in[6];
    const int*   eidx  = (const int*)d_in[7];    // [2][E]
    const int*   edge  = (const int*)d_in[8];    // [EE][2]
    float* out = (float*)d_out;

    char* ws = (char*)d_ws;
    size_t off = 0;
    auto alloc = [&](size_t bytes) -> void* {
        void* p = ws + off;
        off = (off + bytes + 255) & ~(size_t)255;
        return p;
    };
    float*  xbuf = (float*)alloc((size_t)NODES * DIM * sizeof(float));   // 51.2 MB
    float*  hbuf = (float*)alloc((size_t)NODES * DIM * sizeof(float));   // 51.2 MB
    float*  dis  = (float*)alloc((size_t)NODES * sizeof(float));
    __bf16* Wt   = (__bf16*)alloc((size_t)2 * DIM * DIM * sizeof(__bf16));
    __bf16* W1t  = (__bf16*)alloc((size_t)HALF * DIM * sizeof(__bf16));
    (void)ws_size; (void)in_sizes; (void)n_in; (void)out_size;

    const int* srcp = eidx;
    const int* dstp = eidx + NEDGE;

    // degrees -> dis = rsqrt(deg) (0 where deg==0)
    zero_f32<<<256, 256, 0, stream>>>(dis, (size_t)NODES);
    deg_kernel<<<(NEDGE + 255) / 256, 256, 0, stream>>>(dstp, dis, NEDGE);
    rsqrt_kernel<<<(NODES + 255) / 256, 256, 0, stream>>>(dis, NODES);

    // bf16 transposed weights
    convert_weights<<<(2 * DIM * DIM + 255) / 256, 256, 0, stream>>>(convW, W1, Wt, W1t);

    const float* xin = emb;
    for (int l = 0; l < 2; ++l) {
        gemm16_bf16<<<NODES / 16, 256, 0, stream>>>(xin, Wt + (size_t)l * DIM * DIM, hbuf);
        zero_f32<<<2048, 256, 0, stream>>>(xbuf, (size_t)NODES * DIM);
        scatter_msg<<<NEDGE / 8, 256, 0, stream>>>(hbuf, srcp, dstp, dis, xbuf, NEDGE);
        bias_relu<<<(NODES * DIM + 255) / 256, 256, 0, stream>>>(xbuf, convB + l * DIM,
                                                                 NODES * DIM);
        xin = xbuf;
    }

    mlp_kernel<<<NEE / 16, 256, 0, stream>>>(xbuf, edge, W1t, b1, W2, b2, out);
}
